// Chowder_17188459119037
// MI455X (gfx1250) — matrix operand order repaired and checked by reading, not compiled, not run on hardware
//
#include <hip/hip_runtime.h>
#include <hip/hip_bf16.h>

// ---------------------------------------------------------------------------
// Chowder forward: scores = sigmoid(mask_fill(X@W1+b1)) @ W2 + b2,
// top100/bottom100 per slide, then 200->128->64->1 sigmoid MLP.
// Heavy GEMM: v_wmma_f32_16x16x32_bf16 (bf16 in, f32 accum), async-LDS
// double-buffered W1 staging via global_load_async_to_lds_b128 + ASYNCcnt,
// and software-pipelined A-fragment HBM loads (issued one K-chunk ahead).
// ---------------------------------------------------------------------------

#define DIMD   2048
#define DIMH   128
#define NSLIDE 16
#define NTILE  8000
#define NROWS  (NSLIDE * NTILE)   // 128000
#define KCH    64                 // 2048 / 32 K-chunks
#define NT     8                  // 8 column tiles of 16 (H = 128)
#define BPITCH 40                 // padded LDS row pitch (elements)

typedef __attribute__((ext_vector_type(16))) __bf16 v16bf;
typedef __attribute__((ext_vector_type(8)))  __bf16 v8bf;
typedef __attribute__((ext_vector_type(8)))  float  v8f;
typedef __attribute__((ext_vector_type(4)))  int    v4i;

#if defined(__gfx1250__) && __has_builtin(__builtin_amdgcn_global_load_async_to_lds_b128) && __has_builtin(__builtin_amdgcn_s_wait_asynccnt)
#define HAVE_ASYNC_LDS 1
#else
#define HAVE_ASYNC_LDS 0
#endif

// ---------------------------------------------------------------------------
// Kernel 1: pack W1 [2048,128] f32 -> bf16, chunk-major [kc][n=128][k=32]
// so each 32-K chunk is one contiguous 8 KB block.
// ---------------------------------------------------------------------------
__global__ __launch_bounds__(256) void pack_w1_kernel(const float* __restrict__ W1,
                                                      __bf16* __restrict__ W1p) {
  int idx = blockIdx.x * 256 + threadIdx.x;          // 0 .. 262143
  int k   = idx & 31;
  int n   = (idx >> 5) & 127;
  int kc  = idx >> 12;
  W1p[idx] = (__bf16)W1[(kc * 32 + k) * DIMH + n];
}

// ---------------------------------------------------------------------------
// Kernel 2: fused GEMM + bias + mask + sigmoid + W2 dot -> per-row score.
// 256 threads = 8 waves; wave w owns rows [blk*128 + w*16, +16), full H=128.
// ---------------------------------------------------------------------------
__global__ __launch_bounds__(256) void gemm_score_kernel(
    const float* __restrict__ features, const unsigned char* __restrict__ mask,
    const __bf16* __restrict__ W1p, const float* __restrict__ b1,
    const float* __restrict__ W2, const float* __restrict__ b2,
    float* __restrict__ s_top, float* __restrict__ s_bot) {
#if HAVE_ASYNC_LDS
  __shared__ __align__(16) __bf16 Blds[2][DIMH * BPITCH];   // double-buffered
#else
  __shared__ __align__(16) __bf16 Blds[1][DIMH * BPITCH];
#endif

  const int tid  = threadIdx.x;
  const int lane = tid & 31;
  const int hgrp = lane >> 4;        // 0 / 1 (lane half)
  const int l16  = lane & 15;
  const int wave = tid >> 5;

  const long rowBase = (long)blockIdx.x * 128 + wave * 16;
  const float* arow  = features + (size_t)(rowBase + l16) * DIMD;

  v8f acc[NT];
  const v8f zero = {0.f, 0.f, 0.f, 0.f, 0.f, 0.f, 0.f, 0.f};
#pragma unroll
  for (int t = 0; t < NT; ++t) acc[t] = zero;

  // cooperative-load mapping: thread moves 16 bf16 (32 B) of the 8 KB chunk
  const int flat = tid * 16;         // element in [0,4096)
  const int ln   = flat >> 5;        // n row 0..127
  const int lk   = flat & 31;        // 0 or 16

  // A fragment: 16-bit A 16x32 layout.
  // lane half h, element e: K = (e/8)*16 + h*8 + e%8 -> two runs of 8 floats
  auto loadA = [&](int kc, float4& f0, float4& f1, float4& f2, float4& f3) {
    const float* ap = arow + kc * 32 + hgrp * 8;
    f0 = *(const float4*)(ap);
    f1 = *(const float4*)(ap + 4);
    f2 = *(const float4*)(ap + 16);
    f3 = *(const float4*)(ap + 20);
    __builtin_prefetch((const void*)(ap + 64), 0, 1);   // 2 K-chunks ahead
  };

  // ---- one 32-K chunk: cvt staged A regs, LDS B fragments, 8 WMMAs ----
  auto compute = [&](float4 f0, float4 f1, float4 f2, float4 f3,
                     const __bf16* __restrict__ B) {
    v16bf a;
    a[0]  = (__bf16)f0.x; a[1]  = (__bf16)f0.y; a[2]  = (__bf16)f0.z; a[3]  = (__bf16)f0.w;
    a[4]  = (__bf16)f1.x; a[5]  = (__bf16)f1.y; a[6]  = (__bf16)f1.z; a[7]  = (__bf16)f1.w;
    a[8]  = (__bf16)f2.x; a[9]  = (__bf16)f2.y; a[10] = (__bf16)f2.z; a[11] = (__bf16)f2.w;
    a[12] = (__bf16)f3.x; a[13] = (__bf16)f3.y; a[14] = (__bf16)f3.z; a[15] = (__bf16)f3.w;

    v16bf bb[NT];
#pragma unroll
    for (int t = 0; t < NT; ++t) {
      const __bf16* bp = &B[(t * 16 + l16) * BPITCH + hgrp * 8];
      v8bf blo = *(const v8bf*)bp;
      v8bf bhi = *(const v8bf*)(bp + 16);
      bb[t] = __builtin_shufflevector(blo, bhi, 0, 1, 2, 3, 4, 5, 6, 7,
                                      8, 9, 10, 11, 12, 13, 14, 15);
    }
#pragma unroll
    for (int t = 0; t < NT; ++t)
      acc[t] = __builtin_amdgcn_wmma_f32_16x16x32_bf16(
          false, a, false, bb[t], (short)0, acc[t], false, false);
  };

  float4 c0, c1, c2, c3;             // A regs for current chunk
  float4 n0, n1, n2, n3;             // A regs in flight for next chunk

#if HAVE_ASYNC_LDS
  // ---- async global->LDS, double-buffered: fill k+1 while computing k ----
  typedef __attribute__((address_space(1))) v4i* gv4p;   // global int4*
  typedef __attribute__((address_space(3))) v4i* sv4p;   // LDS int4*
  auto fill = [&](int buf, int kc) {
    gv4p src = (gv4p)(W1p + kc * 4096 + flat);
    sv4p dst = (sv4p)&Blds[buf][ln * BPITCH + lk];
    __builtin_amdgcn_global_load_async_to_lds_b128(src, dst, 0, 0);
    __builtin_amdgcn_global_load_async_to_lds_b128(src + 1, dst + 1, 0, 0);
  };

  fill(0, 0);
  loadA(0, c0, c1, c2, c3);
  for (int kc = 0; kc < KCH - 1; ++kc) {
    const int cur = kc & 1;
    __syncthreads();                          // buf cur^1 fully consumed
    fill(cur ^ 1, kc + 1);                    // B chunk k+1 -> other buffer
    loadA(kc + 1, n0, n1, n2, n3);            // A chunk k+1 in flight
    __builtin_amdgcn_s_wait_asynccnt(2);      // older fill (buf cur) landed
    __syncthreads();                          // ... for every wave in the WG
    compute(c0, c1, c2, c3, &Blds[cur][0]);
    c0 = n0; c1 = n1; c2 = n2; c3 = n3;
  }
  __syncthreads();
  __builtin_amdgcn_s_wait_asynccnt(0);
  __syncthreads();
  compute(c0, c1, c2, c3, &Blds[(KCH - 1) & 1][0]);
#else
  // ---- fallback: synchronous VGPR staging, A still pipelined ----
  __bf16* dst = &Blds[0][ln * BPITCH + lk];
  loadA(0, c0, c1, c2, c3);
  for (int kc = 0; kc < KCH; ++kc) {
    __syncthreads();
    const v8bf* src = (const v8bf*)(W1p + kc * 4096 + flat);
    ((v8bf*)dst)[0] = src[0];
    ((v8bf*)dst)[1] = src[1];
    if (kc + 1 < KCH) loadA(kc + 1, n0, n1, n2, n3);
    __syncthreads();
    compute(c0, c1, c2, c3, &Blds[0][0]);
    c0 = n0; c1 = n1; c2 = n2; c3 = n3;
  }
#endif

  // ---- epilogue: h = acc + b1 ; sigmoid ; dot W2 ; + b2 ; mask
  float b1v[NT], w2v[NT];
#pragma unroll
  for (int t = 0; t < NT; ++t) {
    int n = t * 16 + l16;
    b1v[t] = b1[n];
    w2v[t] = W2[n];
  }
  const float b2s = b2[0];

#pragma unroll
  for (int r = 0; r < 8; ++r) {
    // C layout: VGPR r, lane L -> (M = r + 8*(L/16), N = L%16 + 16*t)
    float p = 0.f;
#pragma unroll
    for (int t = 0; t < NT; ++t) {
      float x = acc[t][r] + b1v[t];
      p += w2v[t] / (1.f + __expf(-x));
    }
    p += __shfl_xor(p, 1, 32);
    p += __shfl_xor(p, 2, 32);
    p += __shfl_xor(p, 4, 32);
    p += __shfl_xor(p, 8, 32);   // stays within each 16-lane half
    if (l16 == 0) {
      long row = rowBase + r + 8 * hgrp;
      float s  = p + b2s;
      bool pad = mask[row] != 0;
      s_top[row] = pad ? -__builtin_inff() : s;
      s_bot[row] = pad ?  __builtin_inff() : s;
    }
  }
}

// ---------------------------------------------------------------------------
// Kernel 3: per-slide top-100 (desc) then bottom-100 (asc), exact, ordered.
// ---------------------------------------------------------------------------
__global__ __launch_bounds__(256) void select_kernel(const float* __restrict__ s_top,
                                                     const float* __restrict__ s_bot,
                                                     float* __restrict__ extreme) {
  __shared__ float buf[NTILE];
  __shared__ float rv[256];
  __shared__ int   ri[256];
  const int b = blockIdx.x, tid = threadIdx.x;

  // ---- top 100, descending ----
  for (int i = tid; i < NTILE; i += 256) buf[i] = s_top[b * NTILE + i];
  __syncthreads();
  for (int sel = 0; sel < 100; ++sel) {
    float bv = -__builtin_inff(); int bi = NTILE;
    for (int i = tid; i < NTILE; i += 256) {
      float v = buf[i];
      if (v > bv) { bv = v; bi = i; }
    }
    rv[tid] = bv; ri[tid] = bi;
    __syncthreads();
    for (int s = 128; s > 0; s >>= 1) {
      if (tid < s) {
        float v = rv[tid + s]; int j = ri[tid + s];
        if (v > rv[tid] || (v == rv[tid] && j < ri[tid])) { rv[tid] = v; ri[tid] = j; }
      }
      __syncthreads();
    }
    if (tid == 0) { extreme[b * 200 + sel] = rv[0]; buf[ri[0]] = -__builtin_inff(); }
    __syncthreads();
  }

  // ---- bottom 100, ascending ----
  for (int i = tid; i < NTILE; i += 256) buf[i] = s_bot[b * NTILE + i];
  __syncthreads();
  for (int sel = 0; sel < 100; ++sel) {
    float bv = __builtin_inff(); int bi = NTILE;
    for (int i = tid; i < NTILE; i += 256) {
      float v = buf[i];
      if (v < bv) { bv = v; bi = i; }
    }
    rv[tid] = bv; ri[tid] = bi;
    __syncthreads();
    for (int s = 128; s > 0; s >>= 1) {
      if (tid < s) {
        float v = rv[tid + s]; int j = ri[tid + s];
        if (v < rv[tid] || (v == rv[tid] && j < ri[tid])) { rv[tid] = v; ri[tid] = j; }
      }
      __syncthreads();
    }
    if (tid == 0) { extreme[b * 200 + 100 + sel] = rv[0]; buf[ri[0]] = __builtin_inff(); }
    __syncthreads();
  }
}

// ---------------------------------------------------------------------------
// Kernel 4: tiny prediction MLP 200 -> 128 -> 64 -> 1 (fp32, one block).
// ---------------------------------------------------------------------------
__global__ __launch_bounds__(128) void mlp_kernel(
    const float* __restrict__ extreme,
    const float* __restrict__ Wm1, const float* __restrict__ bm1,
    const float* __restrict__ Wm2, const float* __restrict__ bm2,
    const float* __restrict__ Wm3, const float* __restrict__ bm3,
    float* __restrict__ y) {
  __shared__ float ext[NSLIDE * 200];
  __shared__ float g1[NSLIDE * 128];
  __shared__ float g2[NSLIDE * 64];
  const int tid = threadIdx.x;

  for (int i = tid; i < NSLIDE * 200; i += 128) ext[i] = extreme[i];
  __syncthreads();

  {  // layer 1: 200 -> 128, sigmoid
    int j = tid;
    for (int b = 0; b < NSLIDE; ++b) {
      float s = bm1[j];
      for (int k = 0; k < 200; ++k) s += ext[b * 200 + k] * Wm1[k * 128 + j];
      g1[b * 128 + j] = 1.f / (1.f + __expf(-s));
    }
  }
  __syncthreads();

  if (tid < 64) {  // layer 2: 128 -> 64, sigmoid
    int j = tid;
    for (int b = 0; b < NSLIDE; ++b) {
      float s = bm2[j];
      for (int k = 0; k < 128; ++k) s += g1[b * 128 + k] * Wm2[k * 64 + j];
      g2[b * 64 + j] = 1.f / (1.f + __expf(-s));
    }
  }
  __syncthreads();

  if (tid < NSLIDE) {  // layer 3: 64 -> 1
    int b = tid;
    float s = bm3[0];
    for (int k = 0; k < 64; ++k) s += g2[b * 64 + k] * Wm3[k];
    y[b] = s;
  }
}

// ---------------------------------------------------------------------------
extern "C" void kernel_launch(void* const* d_in, const int* in_sizes, int n_in,
                              void* d_out, int out_size, void* d_ws, size_t ws_size,
                              hipStream_t stream) {
  (void)in_sizes; (void)n_in; (void)out_size; (void)ws_size;
  const float*         features = (const float*)d_in[0];
  const unsigned char* mask     = (const unsigned char*)d_in[1];  // jnp bool_, 1 byte
  const float* W1  = (const float*)d_in[2];
  const float* b1  = (const float*)d_in[3];
  const float* W2  = (const float*)d_in[4];
  const float* b2  = (const float*)d_in[5];
  const float* Wm1 = (const float*)d_in[6];
  const float* bm1 = (const float*)d_in[7];
  const float* Wm2 = (const float*)d_in[8];
  const float* bm2 = (const float*)d_in[9];
  const float* Wm3 = (const float*)d_in[10];
  const float* bm3 = (const float*)d_in[11];

  float* out = (float*)d_out;          // [16] y, then [16*200] extreme_scores
  char*  ws  = (char*)d_ws;
  __bf16* W1p  = (__bf16*)ws;                              // 512 KB
  float*  s_top = (float*)(ws + 524288);                   // 500 KB
  float*  s_bot = (float*)(ws + 524288 + 512000);          // 500 KB

  pack_w1_kernel<<<1024, 256, 0, stream>>>(W1, W1p);
  gemm_score_kernel<<<NROWS / 128, 256, 0, stream>>>(features, mask, W1p, b1, W2,
                                                     b2, s_top, s_bot);
  select_kernel<<<NSLIDE, 256, 0, stream>>>(s_top, s_bot, out + NSLIDE);
  mlp_kernel<<<1, 128, 0, stream>>>(out + NSLIDE, Wm1, bm1, Wm2, bm2, Wm3, bm3, out);
}
